// NetVLAD_73589969650042
// MI455X (gfx1250) — matrix operand order repaired and checked by reading, not compiled, not run on hardware
//
#include <hip/hip_runtime.h>
#include <hip/hip_bf16.h>

// NetVLAD: B=32, D=512, N=1024 (=32x32), K=64
#define B_  32
#define D_  512
#define N_  1024
#define K_  64
#define NC  64          // descriptors per workgroup chunk
#define NCHUNK (N_/NC)  // 16 chunks per batch
#define NT  512         // threads per block = 16 waves (wave32)

typedef __attribute__((ext_vector_type(16))) __bf16 v16bf;
typedef __attribute__((ext_vector_type(8)))  float  v8f;

// float -> bf16 bits, round-to-nearest-even
__device__ __forceinline__ unsigned short f2bf_bits(float f) {
    union { float f; unsigned int u; } v; v.f = f;
    unsigned int u = v.u;
    u += 0x7FFFu + ((u >> 16) & 1u);
    return (unsigned short)(u >> 16);
}

// Element offset (in bf16 units) inside a 16x32 bf16 A-fragment tile
// (512 elements; lane L holds 16 contiguous bf16 at L*16).
// ISA 7.12.2: lanes 0-15: VGPR0-3 = K 0..7, VGPR4-7 = K 16..23 (M = lane)
//             lanes16-31: VGPR0-3 = K 8..15, VGPR4-7 = K 24..31
__device__ __forceinline__ int a_frag_off(int M, int k) {
    int kk   = k & 15;
    int half = (kk >> 3) & 1;
    int L    = half * 16 + M;
    int vg   = ((k & 16) ? 4 : 0) + ((kk & 7) >> 1);
    int j    = vg * 2 + (k & 1);
    return L * 16 + j;
}

// Element offset inside a 32x16 bf16 B-fragment tile (512 elements).
// lanes 0-15 hold K=0..7,16..23 (N = lane), lanes 16-31 hold K=8..15,24..31.
__device__ __forceinline__ int b_frag_off(int kdim, int n) {
    int half = (kdim >> 3) & 1;
    int L    = half * 16 + n;
    int j    = ((kdim & 16) ? 8 : 0) + (kdim & 7);
    return L * 16 + j;
}

// ---------------- kernel 0: zero workspace ----------------
__global__ void nv_zero_ws(float* p, int n) {
    int i = blockIdx.x * blockDim.x + threadIdx.x;
    if (i < n) p[i] = 0.0f;
}

// ---------------- kernel 1: main WMMA kernel ----------------
// LDS layout (dynamic shared, bytes):
//   [0      ,  65536) dA   : d as GEMM1 A-fragments   (64x512 bf16)
//   [65536  , 131072) dB   : d as GEMM2 B-fragments   (64x512 bf16)
//   [131072 , 196608) cw   : conv_w as GEMM1 B-frags  (512x64 bf16)
//   [196608 , 204800) wf   : w as GEMM2 A-fragments   (64x64 bf16)
//   [204800 , 221184) logits (64x64 f32)
//   [221184 , 221440) wsum   (64 f32)
#define SMEM_BYTES 221440

__global__ void nv_main(const float* __restrict__ x,
                        const float* __restrict__ conv_w,
                        float* __restrict__ vscratch,   // B*K*D f32
                        float* __restrict__ wsumg) {    // B*K   f32
    extern __shared__ char smem[];
    unsigned short* dAus = (unsigned short*)(smem);
    unsigned short* dBus = (unsigned short*)(smem + 65536);
    unsigned short* cwus = (unsigned short*)(smem + 131072);
    unsigned short* wfus = (unsigned short*)(smem + 196608);
    float* logitsLDS     = (float*)(smem + 204800);
    float* wsum          = (float*)(smem + 221184);

    const int tid  = threadIdx.x;
    const int wave = tid >> 5;
    const int lane = tid & 31;
    const int b    = blockIdx.x / NCHUNK;
    const int n0   = (blockIdx.x % NCHUNK) * NC;

    // ---- stage descriptors: x[b][dd][n0+nl] -> dA (A-frag) and dB (B-frag), bf16
    for (int e = tid; e < NC * D_; e += NT) {
        int dd = e >> 6;          // 0..511
        int nl = e & 63;          // 0..63  (contiguous in memory -> coalesced)
        float v = x[((size_t)b * D_ + dd) * (size_t)N_ + n0 + nl];
        unsigned short bv = f2bf_bits(v);
        // A-frag for GEMM1: tile (nt = nl/16, dt = dd/32)
        dAus[(((nl >> 4) * 16 + (dd >> 5)) << 9) + a_frag_off(nl & 15, dd & 31)] = bv;
        // B-frag for GEMM2: tile (kt2 = nl/32, dt16 = dd/16)
        dBus[(((nl >> 5) * 32 + (dd >> 4)) << 9) + b_frag_off(nl & 31, dd & 15)] = bv;
    }
    // ---- stage conv_w[k][dd] -> GEMM1 B-fragments
    for (int e = tid; e < K_ * D_; e += NT) {
        int k  = e >> 9;
        int dd = e & 511;         // contiguous -> coalesced
        unsigned short bv = f2bf_bits(conv_w[k * D_ + dd]);
        cwus[(((dd >> 5) * 4 + (k >> 4)) << 9) + b_frag_off(dd & 31, k & 15)] = bv;
    }
    if (tid < K_) wsum[tid] = 0.0f;
    __syncthreads();

    // ---- GEMM1: logits[n][k] = sum_dd d[n][dd] * conv_w[k][dd]
    // 16 waves -> 16 output 16x16 tiles (nt 0..3, kt 0..3), reduce over 16 dt steps
    {
        const int nt = wave >> 2;
        const int kt = wave & 3;
        v8f acc = {};
        #pragma unroll
        for (int dt = 0; dt < 16; ++dt) {
            v16bf a = *(const v16bf*)&dAus[((nt * 16 + dt) << 9) + lane * 16];
            v16bf bb = *(const v16bf*)&cwus[((dt * 4 + kt) << 9) + lane * 16];
            acc = __builtin_amdgcn_wmma_f32_16x16x32_bf16(
                false, a, false, bb, (short)0, acc, false, false);
        }
        // C layout: VGPR r -> M = r + (lane>=16 ? 8 : 0), N = lane%16
        int nrow = nt * 16 + ((lane >= 16) ? 8 : 0);
        int col  = kt * 16 + (lane & 15);
        #pragma unroll
        for (int r = 0; r < 8; ++r)
            logitsLDS[(nrow + r) * K_ + col] = acc[r];
    }
    __syncthreads();

    // ---- softmax over k (one thread per descriptor row); EPS shift is a no-op
    if (tid < NC) {
        float mx = -3.402823466e38f;
        #pragma unroll
        for (int k = 0; k < K_; ++k) mx = fmaxf(mx, logitsLDS[tid * K_ + k]);
        float s = 0.0f;
        #pragma unroll
        for (int k = 0; k < K_; ++k) {
            float e = __expf(logitsLDS[tid * K_ + k] - mx);
            logitsLDS[tid * K_ + k] = e;
            s += e;
        }
        float inv = 1.0f / s;
        #pragma unroll
        for (int k = 0; k < K_; ++k) {
            float w = logitsLDS[tid * K_ + k] * inv;
            atomicAdd(&wsum[k], w);                       // ds_add_f32
            // w^T as GEMM2 A-frag: M = k, Kdim = n ; tile (mt=k/16, kt2=n/32)
            wfus[(((k >> 4) * 2 + (tid >> 5)) << 9) + a_frag_off(k & 15, tid & 31)]
                = f2bf_bits(w);
        }
    }
    __syncthreads();

    // ---- GEMM2: vlad[k][dc] += sum_n w[n][k] * d[n][dc]
    // 128 output 16x16 tiles (mt 0..3 over K, dt16 0..31 over D): 8 tiles per wave
    #pragma unroll
    for (int i = 0; i < 8; ++i) {
        int tileId = wave * 8 + i;
        int mt   = tileId >> 5;     // 0..3
        int dt16 = tileId & 31;     // 0..31
        v8f acc = {};
        #pragma unroll
        for (int kt2 = 0; kt2 < 2; ++kt2) {
            v16bf a = *(const v16bf*)&wfus[((mt * 2 + kt2) << 9) + lane * 16];
            v16bf bb = *(const v16bf*)&dBus[((kt2 * 32 + dt16) << 9) + lane * 16];
            acc = __builtin_amdgcn_wmma_f32_16x16x32_bf16(
                false, a, false, bb, (short)0, acc, false, false);
        }
        int kbase = mt * 16 + ((lane >= 16) ? 8 : 0);
        int dc    = dt16 * 16 + (lane & 15);
        #pragma unroll
        for (int r = 0; r < 8; ++r)
            atomicAdd(&vscratch[((size_t)b * K_ + (kbase + r)) * D_ + dc], acc[r]);
    }

    if (tid < K_) atomicAdd(&wsumg[b * K_ + tid], wsum[tid]);
}

// ---------------- kernel 2: finalize out = vlad - wsum*centroids ----------------
__global__ void nv_finalize(const float* __restrict__ vscratch,
                            const float* __restrict__ wsumg,
                            const float* __restrict__ centroids,
                            float* __restrict__ out) {
    int i = blockIdx.x * blockDim.x + threadIdx.x;
    if (i >= B_ * K_ * D_) return;
    int b   = i / (K_ * D_);
    int rem = i - b * (K_ * D_);
    int k   = rem / D_;
    int dc  = rem - k * D_;
    out[i] = vscratch[i] - wsumg[b * K_ + k] * centroids[k * D_ + dc];
}

extern "C" void kernel_launch(void* const* d_in, const int* in_sizes, int n_in,
                              void* d_out, int out_size, void* d_ws, size_t ws_size,
                              hipStream_t stream) {
    (void)in_sizes; (void)n_in; (void)out_size; (void)ws_size;
    const float* x         = (const float*)d_in[0];   // (B, D, H, W)
    const float* conv_w    = (const float*)d_in[1];   // (K, D)
    const float* centroids = (const float*)d_in[2];   // (K, D)
    float* out = (float*)d_out;                       // (B, K*D)

    float* vscratch = (float*)d_ws;                   // B*K*D
    float* wsumg    = vscratch + (size_t)B_ * K_ * D_;// B*K

    int nz = B_ * K_ * D_ + B_ * K_;
    nv_zero_ws<<<(nz + 255) / 256, 256, 0, stream>>>(vscratch, nz);

    nv_main<<<B_ * NCHUNK, NT, SMEM_BYTES, stream>>>(x, conv_w, vscratch, wsumg);

    int ne = B_ * K_ * D_;
    nv_finalize<<<(ne + 255) / 256, 256, 0, stream>>>(vscratch, wsumg, centroids, out);
}
